// TopKDecoder_33277406609865
// MI455X (gfx1250) — matrix operand order repaired and checked by reading, not compiled, not run on hardware
//
#include <hip/hip_runtime.h>
#include <math.h>

// ---------------------------------------------------------------------------
// Beam-search LSTM decoder for MI455X (gfx1250).
// GEMMs use v_wmma_f32_16x16x32_bf16 (bf16 operands, f32 accumulate).
// If workspace allows (~103MB), weights are pre-packed ONCE per launch into
// bf16 WMMA B-fragment layout -> steady-state inner loop is 2 contiguous
// 32B loads + 1 WMMA, and per-step L2 weight traffic halves (~82MB/step).
// Otherwise falls back to fp32 weight loads with in-register bf16 convert.
// All weights stay L2-resident across the 16 steps (192MB global L2).
// ---------------------------------------------------------------------------

typedef __attribute__((ext_vector_type(16))) __bf16 v16bf;
typedef __attribute__((ext_vector_type(8)))  float  v8f;

#define BK      128          // B*K beam rows
#define HDIM    1024
#define VDIM    32000
#define KBEAM   8
#define BDIM    16
#define TLEN    16
#define G4H     4096
#define SOS_TOK 1
#define EOS_TOK 2
#define UNK_TOK 3
#define KB32    (HDIM / 32)  // number of K=32 chunks
#define NTILE_G (G4H / 16)   // 256 N tiles for gates
#define NTILE_L (VDIM / 16)  // 2000 N tiles for logits
#define CH      64           // top-k scan chunks per batch row
#define CHLEN   ((KBEAM * VDIM) / CH)  // 4000 candidates per chunk

__device__ __forceinline__ __bf16 f2bf(float f) {
  union { float f; unsigned u; } v; v.f = f;
  unsigned u = v.u + 0x7FFFu + ((v.u >> 16) & 1u);   // round to nearest even
  unsigned short s = (unsigned short)(u >> 16);
  union { unsigned short s; __bf16 b; } o; o.s = s; return o.b;
}

// Index into a WMMA-A-fragment packed buffer for element (row r, col k).
//   lanes 0-15  : M=r%16, K = {0..15} of the 32-chunk
//   lanes 16-31 : M=r%16, K = {16..31}
__device__ __forceinline__ int packA_idx(int r, int k) {
  int mt = r >> 4, ml = r & 15;
  int kb = k >> 5, kk = k & 31;
  int lane = ml + ((kk >> 4) << 4);
  int sub  = kk & 15;
  return ((mt * KB32 + kb) * 32 + lane) * 16 + sub;
}

// ---------------------------------------------------------------------------
// One-time weight pack: fp32 [K,N] row-major -> bf16 WMMA B-fragment layout
// [ntile][kblock][lane][16].  One thread builds one 16-element lane chunk:
// reads are lane-coalesced columns, writes are fully contiguous 32B.
// ---------------------------------------------------------------------------
__global__ void k_packB(const float* __restrict__ W, __bf16* __restrict__ out, int N) {
  int idx  = blockIdx.x * blockDim.x + threadIdx.x;  // enumerates (nt,kb,lane)
  int lane = idx & 31;
  int kb   = (idx >> 5) & (KB32 - 1);
  int nt   = idx >> 10;                              // (idx / (32*KB32))
  int ncol  = nt * 16 + (lane & 15);
  int kbase = kb * 32 + ((lane >> 4) << 4);
  v16bf bm;
#pragma unroll
  for (int j = 0; j < 16; ++j) bm[j] = f2bf(W[(kbase + j) * N + ncol]);
  *(v16bf*)(out + (size_t)idx * 16) = bm;
}

// ---------------------------------------------------------------------------
// Init: h = repeat(h0,K) packed to bf16 fragments, c = repeat(c0,K),
// ss = 0 for beam 0 else -inf, tok = SOS.
// ---------------------------------------------------------------------------
__global__ void k_init(const float* __restrict__ h0, const float* __restrict__ c0,
                       float* __restrict__ c, __bf16* __restrict__ hbf,
                       float* __restrict__ ss, int* __restrict__ tok) {
  int idx = blockIdx.x * blockDim.x + threadIdx.x;   // 128*1024 threads
  int r = idx >> 10, j = idx & 1023;
  int bb = r >> 3;
  float hv = h0[bb * HDIM + j];
  c[r * HDIM + j] = c0[bb * HDIM + j];
  hbf[packA_idx(r, j)] = f2bf(hv);
  if (j == 0) { ss[r] = (r & 7) ? -INFINITY : 0.0f; tok[r] = SOS_TOK; }
}

// Embedding gather -> packed bf16 A fragments.
__global__ void k_embed(const float* __restrict__ emb, const int* __restrict__ tok,
                        __bf16* __restrict__ xbf) {
  int idx = blockIdx.x * blockDim.x + threadIdx.x;
  int r = idx >> 10, kcol = idx & 1023;
  float v = emb[tok[r] * HDIM + kcol];
  xbf[packA_idx(r, kcol)] = f2bf(v);
}

// ---------------------------------------------------------------------------
// gates[128,4096] = x@W_ih + h@W_hh + b  -- pre-packed bf16 weight variant
// ---------------------------------------------------------------------------
__global__ void k_gemm_gates_pk(const __bf16* __restrict__ xbf, const __bf16* __restrict__ hbf,
                                const __bf16* __restrict__ WihBf, const __bf16* __restrict__ WhhBf,
                                const float* __restrict__ bias, float* __restrict__ gates) {
  int lane = threadIdx.x & 31;
  int widx = blockIdx.x * (blockDim.x >> 5) + (threadIdx.x >> 5);  // 0..2047
  int nt   = widx & (NTILE_G - 1);
  int mt   = widx >> 8;
  v8f acc = {};
  for (int kb = 0; kb < KB32; ++kb) {
    v16bf a  = *(const v16bf*)(xbf   + ((mt * KB32 + kb) * 32 + lane) * 16);
    v16bf bm = *(const v16bf*)(WihBf + ((nt * KB32 + kb) * 32 + lane) * 16);
    acc = __builtin_amdgcn_wmma_f32_16x16x32_bf16(false, a, false, bm, (short)0, acc, false, false);
  }
  for (int kb = 0; kb < KB32; ++kb) {
    v16bf a  = *(const v16bf*)(hbf   + ((mt * KB32 + kb) * 32 + lane) * 16);
    v16bf bm = *(const v16bf*)(WhhBf + ((nt * KB32 + kb) * 32 + lane) * 16);
    acc = __builtin_amdgcn_wmma_f32_16x16x32_bf16(false, a, false, bm, (short)0, acc, false, false);
  }
  int row0 = mt * 16 + ((lane >> 4) << 3);
  int ncol = nt * 16 + (lane & 15);
  float bv = bias[ncol];
#pragma unroll
  for (int v = 0; v < 8; ++v) gates[(row0 + v) * G4H + ncol] = acc[v] + bv;
}

// Fallback: fp32 weights converted to bf16 in registers.
__global__ void k_gemm_gates_fb(const __bf16* __restrict__ xbf, const __bf16* __restrict__ hbf,
                                const float* __restrict__ Wih, const float* __restrict__ Whh,
                                const float* __restrict__ bias, float* __restrict__ gates) {
  int lane  = threadIdx.x & 31;
  int widx  = blockIdx.x * (blockDim.x >> 5) + (threadIdx.x >> 5);
  int nt    = widx & (NTILE_G - 1);
  int mt    = widx >> 8;
  int ncol  = nt * 16 + (lane & 15);
  int khalf = (lane >> 4) << 4;
  v8f acc = {};
  for (int kb = 0; kb < KB32; ++kb) {
    v16bf a = *(const v16bf*)(xbf + ((mt * KB32 + kb) * 32 + lane) * 16);
    int kbase = kb * 32 + khalf;
    v16bf bm;
#pragma unroll
    for (int j = 0; j < 16; ++j) bm[j] = f2bf(Wih[(kbase + j) * G4H + ncol]);
    acc = __builtin_amdgcn_wmma_f32_16x16x32_bf16(false, a, false, bm, (short)0, acc, false, false);
  }
  for (int kb = 0; kb < KB32; ++kb) {
    v16bf a = *(const v16bf*)(hbf + ((mt * KB32 + kb) * 32 + lane) * 16);
    int kbase = kb * 32 + khalf;
    v16bf bm;
#pragma unroll
    for (int j = 0; j < 16; ++j) bm[j] = f2bf(Whh[(kbase + j) * G4H + ncol]);
    acc = __builtin_amdgcn_wmma_f32_16x16x32_bf16(false, a, false, bm, (short)0, acc, false, false);
  }
  int row0 = mt * 16 + ((lane >> 4) << 3);
  float bv = bias[ncol];
#pragma unroll
  for (int v = 0; v < 8; ++v) gates[(row0 + v) * G4H + ncol] = acc[v] + bv;
}

// ---------------------------------------------------------------------------
// LSTM elementwise: produces h2,c2 (f32) and h2 packed to bf16 fragments.
// ---------------------------------------------------------------------------
__global__ void k_lstm(const float* __restrict__ gates, const float* __restrict__ c,
                       float* __restrict__ h2, float* __restrict__ c2,
                       __bf16* __restrict__ h2bf) {
  int idx = blockIdx.x * blockDim.x + threadIdx.x;
  int r = idx >> 10, j = idx & 1023;
  const float* g = gates + r * G4H;
  float i_ = g[j], f_ = g[HDIM + j], gg = g[2 * HDIM + j], o_ = g[3 * HDIM + j];
  float si = 1.0f / (1.0f + expf(-i_));
  float sf = 1.0f / (1.0f + expf(-f_));
  float so = 1.0f / (1.0f + expf(-o_));
  float cv = sf * c[r * HDIM + j] + si * tanhf(gg);
  float hv = so * tanhf(cv);
  c2[r * HDIM + j] = cv;
  h2[r * HDIM + j] = hv;
  h2bf[packA_idx(r, j)] = f2bf(hv);
}

// ---------------------------------------------------------------------------
// logits[128,32000] = h2@W_out + b_out   (16000 waves -> fills the GPU)
// ---------------------------------------------------------------------------
__global__ void k_gemm_logits_pk(const __bf16* __restrict__ h2bf, const __bf16* __restrict__ WoutBf,
                                 const float* __restrict__ bout, float* __restrict__ logits) {
  int lane = threadIdx.x & 31;
  int widx = blockIdx.x * (blockDim.x >> 5) + (threadIdx.x >> 5);  // 0..15999
  int nt   = widx % NTILE_L;
  int mt   = widx / NTILE_L;
  const __bf16* bp = WoutBf + ((size_t)nt * KB32 * 32) * 16;
  v8f acc = {};
  for (int kb = 0; kb < KB32; ++kb) {
    v16bf a = *(const v16bf*)(h2bf + ((mt * KB32 + kb) * 32 + lane) * 16);
    if (kb + 4 < KB32)                                 // pull ahead from L2
      __builtin_prefetch(bp + ((kb + 4) * 32 + lane) * 16, 0, 3);
    v16bf bm = *(const v16bf*)(bp + (kb * 32 + lane) * 16);
    acc = __builtin_amdgcn_wmma_f32_16x16x32_bf16(false, a, false, bm, (short)0, acc, false, false);
  }
  int row0 = mt * 16 + ((lane >> 4) << 3);
  int ncol = nt * 16 + (lane & 15);
  float bv = bout[ncol];
#pragma unroll
  for (int v = 0; v < 8; ++v) logits[(row0 + v) * VDIM + ncol] = acc[v] + bv;
}

// Fallback: fp32 W_out converted to bf16 in registers.
__global__ void k_gemm_logits_fb(const __bf16* __restrict__ h2bf, const float* __restrict__ Wout,
                                 const float* __restrict__ bout, float* __restrict__ logits) {
  int lane  = threadIdx.x & 31;
  int widx  = blockIdx.x * (blockDim.x >> 5) + (threadIdx.x >> 5);
  int nt    = widx % NTILE_L;
  int mt    = widx / NTILE_L;
  int ncol  = nt * 16 + (lane & 15);
  int khalf = (lane >> 4) << 4;
  v8f acc = {};
  for (int kb = 0; kb < KB32; ++kb) {
    v16bf a = *(const v16bf*)(h2bf + ((mt * KB32 + kb) * 32 + lane) * 16);
    int kbase = kb * 32 + khalf;
    if (kb + 2 < KB32)
      __builtin_prefetch(&Wout[(kbase + 64) * VDIM + ncol], 0, 3);
    v16bf bm;
#pragma unroll
    for (int j = 0; j < 16; ++j) bm[j] = f2bf(Wout[(kbase + j) * VDIM + ncol]);
    acc = __builtin_amdgcn_wmma_f32_16x16x32_bf16(false, a, false, bm, (short)0, acc, false, false);
  }
  int row0 = mt * 16 + ((lane >> 4) << 3);
  float bv = bout[ncol];
#pragma unroll
  for (int v = 0; v < 8; ++v) logits[(row0 + v) * VDIM + ncol] = acc[v] + bv;
}

// Per-row log-sum-exp (max + log(sum exp)) for log_softmax.
__global__ void k_rowstats(const float* __restrict__ logits, float* __restrict__ rowlse) {
  __shared__ float red[256];
  int r = blockIdx.x, tid = threadIdx.x;
  const float* row = logits + r * VDIM;
  float m = -INFINITY;
  for (int j = tid; j < VDIM; j += 256) m = fmaxf(m, row[j]);
  red[tid] = m; __syncthreads();
  for (int off = 128; off > 0; off >>= 1) {
    if (tid < off) red[tid] = fmaxf(red[tid], red[tid + off]);
    __syncthreads();
  }
  m = red[0]; __syncthreads();
  float s = 0.0f;
  for (int j = tid; j < VDIM; j += 256) s += expf(row[j] - m);
  red[tid] = s; __syncthreads();
  for (int off = 128; off > 0; off >>= 1) {
    if (tid < off) red[tid] += red[tid + off];
    __syncthreads();
  }
  if (tid == 0) rowlse[r] = m + logf(red[0]);
}

// ---------------------------------------------------------------------------
// Top-k phase 1: each block scans a chunk of one batch row's 8*32000
// candidates (total = ss + logp, UNK forbidden) and emits its local top-8.
// ---------------------------------------------------------------------------
__global__ void k_topk_scan(const float* __restrict__ logits, const float* __restrict__ rowlse,
                            const float* __restrict__ ss,
                            float* __restrict__ candv, int* __restrict__ candi) {
  __shared__ float sv[2048]; __shared__ int si[2048];
  __shared__ float rv[256];  __shared__ int rs[256];
  int bb = blockIdx.x / CH, ch = blockIdx.x % CH, tid = threadIdx.x;
  int base = ch * CHLEN;
  float tv[8]; int ti[8];
#pragma unroll
  for (int i = 0; i < 8; ++i) { tv[i] = -INFINITY; ti[i] = -1; }
  for (int o = tid; o < CHLEN; o += 256) {
    int cand = base + o;
    int beam = cand / VDIM;
    int col  = cand - beam * VDIM;
    int row  = bb * KBEAM + beam;
    float val = (col == UNK_TOK) ? -INFINITY
                                 : ss[row] + logits[row * VDIM + col] - rowlse[row];
    if (val > tv[7]) {                       // register-resident sorted insert
      int cnt = 0;
#pragma unroll
      for (int q = 0; q < 8; ++q) cnt += (tv[q] >= val) ? 1 : 0;
#pragma unroll
      for (int q = 7; q >= 0; --q) {
        if (q > cnt)       { tv[q] = tv[q - 1]; ti[q] = ti[q - 1]; }
        else if (q == cnt) { tv[q] = val;       ti[q] = cand; }
      }
    }
  }
#pragma unroll
  for (int i = 0; i < 8; ++i) { sv[tid * 8 + i] = tv[i]; si[tid * 8 + i] = ti[i]; }
  __syncthreads();
  for (int sel = 0; sel < 8; ++sel) {        // 8 rounds of block argmax
    float best = -INFINITY; int bslot = -1;
    for (int s = tid; s < 2048; s += 256) {
      float v = sv[s]; if (v > best) { best = v; bslot = s; }
    }
    rv[tid] = best; rs[tid] = bslot; __syncthreads();
    for (int off = 128; off > 0; off >>= 1) {
      if (tid < off && rv[tid + off] > rv[tid]) { rv[tid] = rv[tid + off]; rs[tid] = rs[tid + off]; }
      __syncthreads();
    }
    if (tid == 0) {
      int slot = rs[0], dst = (bb * CH + ch) * 8 + sel;
      candv[dst] = rv[0];
      candi[dst] = (slot >= 0) ? si[slot] : -1;
      if (slot >= 0) sv[slot] = -INFINITY;
    }
    __syncthreads();
  }
}

// ---------------------------------------------------------------------------
// Top-k phase 2: merge CH*8 = 512 candidates per batch row, emit step
// outputs, update ss/tok/pred.
// ---------------------------------------------------------------------------
__global__ void k_topk_final(const float* __restrict__ candv, const int* __restrict__ candi,
                             float* __restrict__ ss, int* __restrict__ tok, int* __restrict__ pred,
                             float* __restrict__ out_scores, int* __restrict__ out_sym,
                             int* __restrict__ out_pred, int t) {
  __shared__ float sv[512]; __shared__ int si[512];
  __shared__ float rv[256]; __shared__ int rs[256];
  __shared__ float wv[8];   __shared__ int wi[8];
  int bb = blockIdx.x, tid = threadIdx.x;
  sv[tid]       = candv[bb * 512 + tid];       si[tid]       = candi[bb * 512 + tid];
  sv[tid + 256] = candv[bb * 512 + 256 + tid]; si[tid + 256] = candi[bb * 512 + 256 + tid];
  __syncthreads();
  for (int sel = 0; sel < 8; ++sel) {
    float best = sv[tid]; int bslot = tid;
    float v2 = sv[tid + 256]; if (v2 > best) { best = v2; bslot = tid + 256; }
    rv[tid] = best; rs[tid] = bslot; __syncthreads();
    for (int off = 128; off > 0; off >>= 1) {
      if (tid < off && rv[tid + off] > rv[tid]) { rv[tid] = rv[tid + off]; rs[tid] = rs[tid + off]; }
      __syncthreads();
    }
    if (tid == 0) {
      int slot = rs[0];
      wv[sel] = rv[0];
      int ci = (slot >= 0) ? si[slot] : -1;
      wi[sel] = (ci >= 0) ? ci : 0;
      if (slot >= 0) sv[slot] = -INFINITY;
    }
    __syncthreads();
  }
  if (tid < 8) {
    int k = tid;
    float val = wv[k];
    int cand  = wi[k];
    int predl = cand / VDIM;
    int sym   = cand - predl * VDIM;
    int row   = bb * KBEAM + k;
    int outo  = (t * BDIM + bb) * KBEAM + k;
    out_scores[outo] = val;          // pre-mask score, as in reference
    out_sym[outo]    = sym;
    out_pred[outo]   = predl;
    ss[row]  = (sym == EOS_TOK) ? -INFINITY : val;
    tok[row] = sym;
    pred[row] = bb * KBEAM + predl;  // flat predecessor beam index
  }
}

// Beam gather: c <- c2[pred], hbf <- pack(h2[pred]) for next step.
__global__ void k_gather(const float* __restrict__ h2, const float* __restrict__ c2,
                         const int* __restrict__ pred,
                         float* __restrict__ c, __bf16* __restrict__ hbf) {
  int idx = blockIdx.x * blockDim.x + threadIdx.x;
  int r = idx >> 10, j = idx & 1023;
  int p = pred[r];
  float hv = h2[p * HDIM + j];
  c[r * HDIM + j] = c2[p * HDIM + j];
  hbf[packA_idx(r, j)] = f2bf(hv);
}

// ---------------------------------------------------------------------------
extern "C" void kernel_launch(void* const* d_in, const int* in_sizes, int n_in,
                              void* d_out, int out_size, void* d_ws, size_t ws_size,
                              hipStream_t stream) {
  (void)in_sizes; (void)n_in; (void)out_size;
  const float* embedding = (const float*)d_in[0];
  const float* W_ih  = (const float*)d_in[1];
  const float* W_hh  = (const float*)d_in[2];
  const float* bvec  = (const float*)d_in[3];
  const float* W_out = (const float*)d_in[4];
  const float* b_out = (const float*)d_in[5];
  const float* h0    = (const float*)d_in[6];
  const float* c0    = (const float*)d_in[7];

  char* ws = (char*)d_ws;
  size_t off = 0;
  auto alloc = [&](size_t bytes) -> void* {
    void* p = ws + off;
    off = (off + bytes + 255) & ~(size_t)255;
    return p;
  };
  float*  c_st   = (float*)alloc((size_t)BK * HDIM * 4);
  float*  c2_st  = (float*)alloc((size_t)BK * HDIM * 4);
  float*  h2_st  = (float*)alloc((size_t)BK * HDIM * 4);
  float*  gates  = (float*)alloc((size_t)BK * G4H * 4);
  float*  logits = (float*)alloc((size_t)BK * VDIM * 4);
  float*  rowlse = (float*)alloc(BK * 4);
  float*  ss     = (float*)alloc(BK * 4);
  int*    tok    = (int*)alloc(BK * 4);
  int*    pred   = (int*)alloc(BK * 4);
  __bf16* xbf    = (__bf16*)alloc((size_t)BK * HDIM * 2);
  __bf16* hbf    = (__bf16*)alloc((size_t)BK * HDIM * 2);
  __bf16* h2bf   = (__bf16*)alloc((size_t)BK * HDIM * 2);
  float*  candv  = (float*)alloc((size_t)BDIM * CH * 8 * 4);
  int*    candi  = (int*)alloc((size_t)BDIM * CH * 8 * 4);
  // Optional pre-packed bf16 weights (used only if workspace is big enough).
  __bf16* WihBf  = (__bf16*)alloc((size_t)HDIM * G4H * 2);
  __bf16* WhhBf  = (__bf16*)alloc((size_t)HDIM * G4H * 2);
  __bf16* WoutBf = (__bf16*)alloc((size_t)HDIM * VDIM * 2);
  const bool prepack = (ws_size >= off);   // fixed per run -> deterministic

  float* out_scores = (float*)d_out;                       // [T,B,K] f32
  int*   out_sym    = (int*)d_out + TLEN * BDIM * KBEAM;   // [T,B,K] i32
  int*   out_pred   = (int*)d_out + 2 * TLEN * BDIM * KBEAM;

  if (prepack) {  // one-time fp32 -> bf16 fragment pack (amortized over T steps)
    k_packB<<<(HDIM * G4H / 16) / 256, 256, 0, stream>>>(W_ih, WihBf, G4H);
    k_packB<<<(HDIM * G4H / 16) / 256, 256, 0, stream>>>(W_hh, WhhBf, G4H);
    k_packB<<<(HDIM * VDIM / 16) / 256, 256, 0, stream>>>(W_out, WoutBf, VDIM);
  }
  k_init<<<512, 256, 0, stream>>>(h0, c0, c_st, hbf, ss, tok);
  for (int t = 0; t < TLEN; ++t) {
    k_embed<<<512, 256, 0, stream>>>(embedding, tok, xbf);
    if (prepack)
      k_gemm_gates_pk<<<256, 256, 0, stream>>>(xbf, hbf, WihBf, WhhBf, bvec, gates);
    else
      k_gemm_gates_fb<<<256, 256, 0, stream>>>(xbf, hbf, W_ih, W_hh, bvec, gates);
    k_lstm<<<512, 256, 0, stream>>>(gates, c_st, h2_st, c2_st, h2bf);
    if (prepack)
      k_gemm_logits_pk<<<2000, 256, 0, stream>>>(h2bf, WoutBf, b_out, logits);
    else
      k_gemm_logits_fb<<<2000, 256, 0, stream>>>(h2bf, W_out, b_out, logits);
    k_rowstats<<<BK, 256, 0, stream>>>(logits, rowlse);
    k_topk_scan<<<BDIM * CH, 256, 0, stream>>>(logits, rowlse, ss, candv, candi);
    k_topk_final<<<BDIM, 256, 0, stream>>>(candv, candi, ss, tok, pred,
                                           out_scores, out_sym, out_pred, t);
    k_gather<<<512, 256, 0, stream>>>(h2_st, c2_st, pred, c_st, hbf);
  }
}
